// FullAttention_73555609911677
// MI455X (gfx1250) — matrix operand order repaired and checked by reading, not compiled, not run on hardware
//
#include <hip/hip_runtime.h>
#include <math.h>

typedef __attribute__((ext_vector_type(16))) _Float16 v16h;
typedef __attribute__((ext_vector_type(8)))  float    v8f;

#define Bb 4
#define Ll 2048
#define Hh 8
#define Ee 64
#define BM 64          // query rows per workgroup (16 per wave, 4 waves)
#define BN 64          // keys per iteration
#define LDT 72         // padded LDS row stride (halves) for K^T / V tiles
#define LDP 66         // padded LDS row stride (halves) for P scratch

// XOR-butterfly within 16-lane rows, on the VALU via DPP16 ROW_XMASK (0x160|m).
// Keeps reductions off the DS pipe (which feeds the WMMA fragments).
template <int MASK>
__device__ __forceinline__ float dpp_xor16(float v) {
  int s = __float_as_int(v);
  int d = __builtin_amdgcn_update_dpp(s, s, 0x160 | MASK, 0xF, 0xF, false);
  return __int_as_float(d);
}
__device__ __forceinline__ float rowmax16(float v) {
  v = fmaxf(v, dpp_xor16<1>(v));
  v = fmaxf(v, dpp_xor16<2>(v));
  v = fmaxf(v, dpp_xor16<4>(v));
  v = fmaxf(v, dpp_xor16<8>(v));
  return v;
}
__device__ __forceinline__ float rowsum16(float v) {
  v += dpp_xor16<1>(v);
  v += dpp_xor16<2>(v);
  v += dpp_xor16<4>(v);
  v += dpp_xor16<8>(v);
  return v;
}

__global__ __launch_bounds__(128) void fattn_fwd_kernel(
    const float* __restrict__ Qg, const float* __restrict__ Kg,
    const float* __restrict__ Vg, float* __restrict__ Og)
{
  __shared__ _Float16 sKt[Ee * LDT];          // K tile transposed: [e][key]
  __shared__ _Float16 sV [BN * LDT];          // V tile: [key][e]
  __shared__ _Float16 sP [4 * 16 * LDP];      // per-wave P scratch: [16 rows][BN keys]

  const int tid  = threadIdx.x;
  const int wave = tid >> 5;
  const int lane = tid & 31;
  const int m16  = lane & 15;   // A-row / C-column index
  const int hi   = lane >> 4;   // lane half select

  const int qblk = blockIdx.x & (Ll / BM - 1);        // 32 q-blocks
  const int bh   = blockIdx.x >> 5;
  const int h    = bh & (Hh - 1);
  const int b    = bh >> 3;

  const int q0 = qblk * BM;

  // ---- Q fragments (A layout, 16-bit 16x32): lane m16 = row M; hi selects K-half
  // Fold softmax scale AND log2(e) into Q so all exponentials are bare v_exp_f32:
  // softmax_e(s/sqrt(E)) == softmax_2(s * log2e/sqrt(E)).
  const float scale = 0.125f * 1.44269504088896340736f;
  const float* qp = Qg + (((size_t)b * Ll + (q0 + wave * 16 + m16)) * Hh + h) * Ee;
  v16h qa0, qa1;
#pragma unroll
  for (int vg = 0; vg < 8; ++vg) {
    int k0 = (vg < 4) ? (8 * hi + 2 * vg) : (16 + 8 * hi + 2 * (vg - 4));
    qa0[2 * vg]     = (_Float16)(qp[k0]          * scale);
    qa0[2 * vg + 1] = (_Float16)(qp[k0 + 1]      * scale);
    qa1[2 * vg]     = (_Float16)(qp[32 + k0]     * scale);
    qa1[2 * vg + 1] = (_Float16)(qp[32 + k0 + 1] * scale);
  }

  // Output accumulators: 4 N-tiles (E=64) in C layout; per-lane row stats (base-2).
  v8f o[4];
  float mrow[8], lrow[8];
#pragma unroll
  for (int nt = 0; nt < 4; ++nt) o[nt] = (v8f){0,0,0,0,0,0,0,0};
#pragma unroll
  for (int r = 0; r < 8; ++r) { mrow[r] = -INFINITY; lrow[r] = 0.0f; }

  _Float16* pw = &sP[wave * 16 * LDP];

  for (int kb = 0; kb <= qblk; ++kb) {
    const int kg0 = kb * BN;

    // ---- stage K (transposed) and V tiles into LDS as f16; 2048 float2 / 128 thr
#pragma unroll
    for (int it = 0; it < (BN * Ee / 2) / 128; ++it) {
      int i2  = (it * 128 + tid) * 2;
      int row = i2 >> 6;          // key index 0..63
      int col = i2 & 63;          // feature index (even)
      const size_t gofs = (((size_t)b * Ll + (kg0 + row)) * Hh + h) * Ee + col;
      float k0f = Kg[gofs], k1f = Kg[gofs + 1];
      float v0f = Vg[gofs], v1f = Vg[gofs + 1];
      sKt[(col    ) * LDT + row] = (_Float16)k0f;
      sKt[(col + 1) * LDT + row] = (_Float16)k1f;
      sV [row * LDT + col    ]   = (_Float16)v0f;
      sV [row * LDT + col + 1]   = (_Float16)v1f;
    }
    // prefetch next tile while we compute on this one
    if (kb < qblk) {
      const size_t nofs = (((size_t)b * Ll + (kg0 + BN + (tid >> 1))) * Hh + h) * Ee
                          + (tid & 1) * 32;
      __builtin_prefetch(Kg + nofs, 0, 0);
      __builtin_prefetch(Vg + nofs, 0, 0);
    }
    __syncthreads();

    // ---- S = (Q*scale) @ K^T  : 4 key N-tiles x 2 K-chunks of 32
    v8f s[4];
#pragma unroll
    for (int nt = 0; nt < 4; ++nt) {
      v8f acc = (v8f){0,0,0,0,0,0,0,0};
#pragma unroll
      for (int ec = 0; ec < 2; ++ec) {
        // B layout (32x16 f16): lane m16 = N (key), element j = K (e, table order by hi)
        v16h bf;
        const _Float16* kc = &sKt[(ec * 32) * LDT + nt * 16 + m16];
#pragma unroll
        for (int j = 0; j < 16; ++j) bf[j] = kc[((j < 8) ? (8 * hi + j) : (16 + 8 * hi + (j - 8))) * LDT];
        acc = __builtin_amdgcn_wmma_f32_16x16x32_f16(
            false, ec ? qa1 : qa0, false, bf, (short)0, acc, false, false);
      }
      s[nt] = acc;
    }

    // ---- causal mask on the diagonal block (C layout: M = r + 8*hi, N = m16)
    if (kb == qblk) {
#pragma unroll
      for (int nt = 0; nt < 4; ++nt)
#pragma unroll
        for (int r = 0; r < 8; ++r) {
          int qi = wave * 16 + r + 8 * hi;   // relative row (q0 == kg0)
          int kj = nt * 16 + m16;            // relative key
          s[nt][r] = (kj > qi) ? -INFINITY : s[nt][r];
        }
    }

    // ---- online softmax (base-2): row max over 64 keys via DPP butterfly
    float corr[8];
#pragma unroll
    for (int r = 0; r < 8; ++r) {
      float mx = rowmax16(fmaxf(fmaxf(s[0][r], s[1][r]), fmaxf(s[2][r], s[3][r])));
      float mnew = fmaxf(mrow[r], mx);
      corr[r] = __builtin_amdgcn_exp2f(mrow[r] - mnew);  // 0 on first block (m=-inf)
      mrow[r] = mnew;
    }

    // ---- P = 2^(S - m); accumulate row sums; stash P (f16) in per-wave LDS
    float rs[8];
#pragma unroll
    for (int r = 0; r < 8; ++r) rs[r] = 0.0f;
#pragma unroll
    for (int nt = 0; nt < 4; ++nt)
#pragma unroll
      for (int r = 0; r < 8; ++r) {
        float p = __builtin_amdgcn_exp2f(s[nt][r] - mrow[r]);  // 2^-inf = 0 (mask)
        rs[r] += p;
        pw[(r + 8 * hi) * LDP + nt * 16 + m16] = (_Float16)p;
      }
#pragma unroll
    for (int r = 0; r < 8; ++r) {
      float t = rowsum16(rs[r]);
      lrow[r] = lrow[r] * corr[r] + t;
      // rescale running output rows
#pragma unroll
      for (int nt = 0; nt < 4; ++nt) o[nt][r] *= corr[r];
    }

    // wave-private LDS RAW: DS ops are in-order per wave; block compiler reordering
    asm volatile("s_wait_dscnt 0" ::: "memory");

    // ---- O += P @ V : 2 K-chunks of 32 keys x 4 feature N-tiles
#pragma unroll
    for (int kc = 0; kc < 2; ++kc) {
      v16h af;   // A layout: lane m16 = query row, K = key index
#pragma unroll
      for (int vg = 0; vg < 8; ++vg) {
        int k0 = kc * 32 + ((vg < 4) ? (8 * hi + 2 * vg)
                                     : (16 + 8 * hi + 2 * (vg - 4)));
        af[2 * vg]     = pw[m16 * LDP + k0];
        af[2 * vg + 1] = pw[m16 * LDP + k0 + 1];
      }
#pragma unroll
      for (int nt = 0; nt < 4; ++nt) {
        v16h bf;   // B layout: lane m16 = feature, element j = key (table order by hi)
        const _Float16* vr = &sV[(kc * 32) * LDT + nt * 16 + m16];
#pragma unroll
        for (int j = 0; j < 16; ++j) bf[j] = vr[((j < 8) ? (8 * hi + j) : (16 + 8 * hi + (j - 8))) * LDT];
        o[nt] = __builtin_amdgcn_wmma_f32_16x16x32_f16(
            false, af, false, bf, (short)0, o[nt], false, false);
      }
    }
    __syncthreads();   // protect sKt/sV before next iteration's staging
  }

  // ---- normalize and write out [B, L, H, E]
#pragma unroll
  for (int r = 0; r < 8; ++r) {
    float inv = 1.0f / lrow[r];
    int qi = q0 + wave * 16 + r + 8 * hi;
    float* op = Og + (((size_t)b * Ll + qi) * Hh + h) * Ee;
#pragma unroll
    for (int nt = 0; nt < 4; ++nt)
      op[nt * 16 + m16] = o[nt][r] * inv;
  }
}

extern "C" void kernel_launch(void* const* d_in, const int* in_sizes, int n_in,
                              void* d_out, int out_size, void* d_ws, size_t ws_size,
                              hipStream_t stream) {
  (void)in_sizes; (void)n_in; (void)out_size; (void)d_ws; (void)ws_size;
  const float* Q = (const float*)d_in[0];
  const float* K = (const float*)d_in[1];
  const float* V = (const float*)d_in[2];
  float* O = (float*)d_out;
  dim3 grid(Bb * Hh * (Ll / BM));   // 1024 workgroups, 4 waves each
  dim3 block(128);
  fattn_fwd_kernel<<<grid, block, 0, stream>>>(Q, K, V, O);
}
